// PAM_68676527063164
// MI455X (gfx1250) — matrix operand order repaired and checked by reading, not compiled, not run on hardware
//
#include <hip/hip_runtime.h>

typedef unsigned short ushort_t;

typedef __attribute__((ext_vector_type(16))) __bf16 v16bf;
typedef __attribute__((ext_vector_type(8)))  __bf16 v8bf;
typedef __attribute__((ext_vector_type(8)))  float  v8f;

#define BS 4
#define NF 512
#define C8 64
#define HW 4096
#define LOG2E 1.4426950408889634f

#if __has_builtin(__builtin_amdgcn_exp2f)
#define EXP2(x) __builtin_amdgcn_exp2f(x)
#else
#define EXP2(x) exp2f(x)
#endif

static __device__ __forceinline__ float bf2f(ushort_t h) {
  union { unsigned u; float f; } v; v.u = ((unsigned)h) << 16;
  return v.f;
}
// f32 -> bf16 (RNE) in one VALU op via V_FMA_MIXLO_BF16 (CDNA5 VOP3P op 62)
static __device__ __forceinline__ ushort_t f2bf(float f) {
  unsigned d;
  asm("v_fma_mixlo_bf16 %0, %1, 1.0, 0" : "=v"(d) : "v"(f));
  return (ushort_t)d;
}
// pack two f32 -> dword {lo: bf16(a), hi: bf16(b)} via mixlo + mixhi
static __device__ __forceinline__ unsigned pk2bf(float a, float b) {
  unsigned d;
  asm("v_fma_mixlo_bf16 %0, %1, 1.0, 0" : "=v"(d) : "v"(a));
  asm("v_fma_mixhi_bf16 %0, %1, 1.0, 0" : "+v"(d) : "v"(b));
  return d;
}
static __device__ __forceinline__ v8f wmma_bf16(v16bf a, v16bf b, v8f c) {
  // (neg_a, A, neg_b, B, c_mod, C, reuse_a, reuse_b)
  return __builtin_amdgcn_wmma_f32_16x16x32_bf16(false, a, false, b, (short)0, c,
                                                 false, false);
}
static __device__ __forceinline__ v16bf cat8(v8bf lo, v8bf hi) {
  return __builtin_shufflevector(lo, hi, 0,1,2,3,4,5,6,7,8,9,10,11,12,13,14,15);
}

// ---------------------------------------------------------------------------
// Projection for q and k: y[b][o][n] = sum_c W[o][c] x[b][c][n] + bias[o],
// stored transposed as qt/kt[b][n][o] in bf16 (channel-contiguous rows for WMMA).
// grid (16 n-blocks, 8 o-groups [0..3]=q [4..7]=k, 4 batches), 256 threads.
// ---------------------------------------------------------------------------
__global__ void pam_proj_qk(const float* __restrict__ x,
                            const float* __restrict__ Wq, const float* __restrict__ bq,
                            const float* __restrict__ Wk, const float* __restrict__ bk,
                            ushort_t* __restrict__ qt, ushort_t* __restrict__ kt) {
  __shared__ float wt[16 * NF];
  const int t  = threadIdx.x;
  const int nb = blockIdx.x;
  const int og = blockIdx.y;
  const int b  = blockIdx.z;
  const bool isq = (og < 4);
  const int o0 = (isq ? og : og - 4) * 16;
  const float* W    = isq ? Wq : Wk;
  const float* bias = isq ? bq : bk;
  ushort_t* dst     = isq ? qt : kt;

  for (int i = t; i < 16 * NF; i += 256) wt[i] = W[o0 * NF + i];
  __syncthreads();

  const int n = nb * 256 + t;
  const float* xb = x + (size_t)b * NF * HW + n;
  float acc[16];
#pragma unroll
  for (int u = 0; u < 16; ++u) acc[u] = 0.f;
  for (int c = 0; c < NF; ++c) {
    float xv = xb[(size_t)c * HW];
#pragma unroll
    for (int u = 0; u < 16; ++u) acc[u] = fmaf(wt[u * NF + c], xv, acc[u]);
  }
  ushort_t* drow = dst + ((size_t)b * HW + n) * C8 + o0;
#pragma unroll
  for (int u = 0; u < 16; ++u) drow[u] = f2bf(acc[u] + bias[o0 + u]);
}

// ---------------------------------------------------------------------------
// Projection for v: vb[b][o][n] = sum_c Wv[o][c] x[b][c][n] + bv[o]  (bf16).
// Bias folded in is exact since softmax rows sum to 1.
// grid (16 n-blocks, 32 o-groups, 4 batches), 256 threads.
// ---------------------------------------------------------------------------
__global__ void pam_proj_v(const float* __restrict__ x,
                           const float* __restrict__ Wv, const float* __restrict__ bv,
                           ushort_t* __restrict__ vb) {
  __shared__ float wt[16 * NF];
  const int t  = threadIdx.x;
  const int nb = blockIdx.x;
  const int o0 = blockIdx.y * 16;
  const int b  = blockIdx.z;

  for (int i = t; i < 16 * NF; i += 256) wt[i] = Wv[o0 * NF + i];
  __syncthreads();

  const int n = nb * 256 + t;
  const float* xb = x + (size_t)b * NF * HW + n;
  float acc[16];
#pragma unroll
  for (int u = 0; u < 16; ++u) acc[u] = 0.f;
  for (int c = 0; c < NF; ++c) {
    float xv = xb[(size_t)c * HW];
#pragma unroll
    for (int u = 0; u < 16; ++u) acc[u] = fmaf(wt[u * NF + c], xv, acc[u]);
  }
#pragma unroll
  for (int u = 0; u < 16; ++u)
    vb[((size_t)b * NF + o0 + u) * HW + n] = f2bf(acc[u] + bv[o0 + u]);
}

// ---------------------------------------------------------------------------
// Pass A: softmax statistics. One workgroup per (b, i): m_i = max_j S[i,j],
// rinv_i = 1 / sum_j exp(S[i,j]-m_i).  S[i,j] = sum_c kt[i][c]*qt[j][c].
// grid (4096, 4), 256 threads.
// ---------------------------------------------------------------------------
__global__ void pam_stats(const ushort_t* __restrict__ qt,
                          const ushort_t* __restrict__ kt,
                          float* __restrict__ mrow, float* __restrict__ rinv) {
  __shared__ float krow[C8];
  __shared__ float red[256];
  const int t = threadIdx.x;
  const int i = blockIdx.x;
  const int b = blockIdx.y;

  const ushort_t* kp = kt + ((size_t)b * HW + i) * C8;
  if (t < C8) krow[t] = bf2f(kp[t]);
  __syncthreads();

  float kr[C8];
#pragma unroll
  for (int c = 0; c < C8; ++c) kr[c] = krow[c];

  const ushort_t* qbase = qt + (size_t)b * HW * C8;
  float s[16];
  float m = -3.402823e38f;
#pragma unroll
  for (int jj = 0; jj < 16; ++jj) {
    const int j = jj * 256 + t;
    const uint4* q4 = (const uint4*)(qbase + (size_t)j * C8);
    float acc = 0.f;
#pragma unroll
    for (int c4 = 0; c4 < 8; ++c4) {
      uint4 w = q4[c4];
      acc = fmaf(kr[c4*8+0], bf2f((ushort_t)(w.x & 0xffff)), acc);
      acc = fmaf(kr[c4*8+1], bf2f((ushort_t)(w.x >> 16)),    acc);
      acc = fmaf(kr[c4*8+2], bf2f((ushort_t)(w.y & 0xffff)), acc);
      acc = fmaf(kr[c4*8+3], bf2f((ushort_t)(w.y >> 16)),    acc);
      acc = fmaf(kr[c4*8+4], bf2f((ushort_t)(w.z & 0xffff)), acc);
      acc = fmaf(kr[c4*8+5], bf2f((ushort_t)(w.z >> 16)),    acc);
      acc = fmaf(kr[c4*8+6], bf2f((ushort_t)(w.w & 0xffff)), acc);
      acc = fmaf(kr[c4*8+7], bf2f((ushort_t)(w.w >> 16)),    acc);
    }
    s[jj] = acc;
    m = fmaxf(m, acc);
  }

  red[t] = m;
  __syncthreads();
  for (int off = 128; off > 0; off >>= 1) {
    if (t < off) red[t] = fmaxf(red[t], red[t + off]);
    __syncthreads();
  }
  const float M = red[0];
  __syncthreads();

  float lsum = 0.f;
  const float mneg = -M * LOG2E;
#pragma unroll
  for (int jj = 0; jj < 16; ++jj) lsum += EXP2(fmaf(s[jj], LOG2E, mneg));
  red[t] = lsum;
  __syncthreads();
  for (int off = 128; off > 0; off >>= 1) {
    if (t < off) red[t] += red[t + off];
    __syncthreads();
  }
  if (t == 0) {
    mrow[(size_t)b * HW + i] = M;
    rinv[(size_t)b * HW + i] = 1.0f / red[0];
  }
}

// ---------------------------------------------------------------------------
// Pass B: fused scores -> exp -> P·V -> residual. One workgroup per
// (b, 16-row i-tile); 8 waves; wave w owns output channels [64w, 64w+64).
// Per j-chunk of 32: 4 WMMA for S (K=64 in 2 steps, two 16-col subtiles),
// exp2(fma) + v_fma_mixlo/hi_bf16 pack + LDS transpose to A-layout,
// 4 WMMA for P·V.  grid (256, 4), 256 threads.
// ---------------------------------------------------------------------------
__global__ void pam_attn_out(const ushort_t* __restrict__ qt,
                             const ushort_t* __restrict__ kt,
                             const ushort_t* __restrict__ vb,
                             const float* __restrict__ mrow,
                             const float* __restrict__ rinv,
                             const float* __restrict__ x,
                             const float* __restrict__ alpha,
                             float* __restrict__ out) {
  __shared__ ushort_t pbuf[8][16 * 32];   // per-wave P tile (row-major 16 x 32)
  const int lane = threadIdx.x & 31;
  const int wave = threadIdx.x >> 5;
  const int hi   = lane >> 4;      // lane half
  const int ln   = lane & 15;
  const int i0 = blockIdx.x * 16;
  const int b  = blockIdx.y;
  ushort_t* pw = &pbuf[wave][0];

  // A-matrix (Kᵀ tile, 16 i-rows x 64 channels) — stays resident.
  // A layout: lane l holds row M=l%16; element e -> K = cbase + (l/16)*16 + e
  const ushort_t* krow = kt + ((size_t)b * HW + (i0 + ln)) * C8;
  const v16bf ak0 = *(const v16bf*)(krow +      hi * 16);
  const v16bf ak1 = *(const v16bf*)(krow + 32 + hi * 16);

  // per-lane row maxima for the 8 rows this lane accumulates (M = r + 8*hi),
  // pre-scaled so p = exp2(fma(S, log2e, mneg))
  const float* mp = mrow + (size_t)b * HW + i0 + hi * 8;
  float mneg[8];
#pragma unroll
  for (int r = 0; r < 8; ++r) mneg[r] = -mp[r] * LOG2E;

  v8f O[4];
#pragma unroll
  for (int tt = 0; tt < 4; ++tt) O[tt] = (v8f){0.f,0.f,0.f,0.f,0.f,0.f,0.f,0.f};

  const ushort_t* qbase = qt + (size_t)b * HW * C8;
  const ushort_t* vbase = vb + (size_t)b * NF * HW;

  for (int j0 = 0; j0 < HW; j0 += 32) {
    // B-matrices for Q: lane l -> col N=l%16, element e -> K = cbase + (l/16)*16 + e
    const ushort_t* qr0 = qbase + (size_t)(j0 +      ln) * C8;
    const ushort_t* qr1 = qbase + (size_t)(j0 + 16 + ln) * C8;
    const v16bf bq00 = *(const v16bf*)(qr0 +      hi * 16);
    const v16bf bq01 = *(const v16bf*)(qr0 + 32 + hi * 16);
    const v16bf bq10 = *(const v16bf*)(qr1 +      hi * 16);
    const v16bf bq11 = *(const v16bf*)(qr1 + 32 + hi * 16);

    v8f S0 = (v8f){0.f,0.f,0.f,0.f,0.f,0.f,0.f,0.f};
    v8f S1 = (v8f){0.f,0.f,0.f,0.f,0.f,0.f,0.f,0.f};
    S0 = wmma_bf16(ak0, bq00, S0);
    S0 = wmma_bf16(ak1, bq01, S0);
    S1 = wmma_bf16(ak0, bq10, S1);
    S1 = wmma_bf16(ak1, bq11, S1);

    // p = exp2(S*log2e - m*log2e); pack pair with fma_mixlo/mixhi_bf16;
    // C-layout -> row-major LDS tile (rows M = r+8*hi, cols ln / ln+16)
#pragma unroll
    for (int r = 0; r < 8; ++r) {
      const float p0 = EXP2(fmaf(S0[r], LOG2E, mneg[r]));
      const float p1 = EXP2(fmaf(S1[r], LOG2E, mneg[r]));
      const unsigned pk = pk2bf(p0, p1);
      const int row = (r + 8 * hi) * 32;
      pw[row + ln]      = (ushort_t)pk;          // ds_store_b16 (low half)
      pw[row + ln + 16] = (ushort_t)(pk >> 16);  // ds_store_b16_d16_hi
    }
    asm volatile("s_wait_dscnt 0x0" ::: "memory");

    // reload P in A-layout: lane l -> row l%16, K = (l/16)*16 + e
    const v16bf aP = *(const v16bf*)(pw + ln * 32 + hi * 16);

    // P·V over this wave's 4 channel tiles (+ prefetch next chunks of V)
#pragma unroll
    for (int tt = 0; tt < 4; ++tt) {
      const int c = wave * 64 + tt * 16 + ln;
      const ushort_t* vr = vbase + (size_t)c * HW + j0 + hi * 16;
      __builtin_prefetch(vr + 128, 0, 0);        // ~256B ahead in the j stream
      const v16bf bV = *(const v16bf*)vr;
      O[tt] = wmma_bf16(aP, bV, O[tt]);
    }
  }

  // normalize by 1/l_i, fuse residual x + alpha*out
  const float* rp = rinv + (size_t)b * HW + i0 + hi * 8;
  float rv[8];
#pragma unroll
  for (int r = 0; r < 8; ++r) rv[r] = rp[r];
  const float al = alpha[0];

#pragma unroll
  for (int tt = 0; tt < 4; ++tt) {
    const int c = wave * 64 + tt * 16 + ln;
    const size_t base = ((size_t)b * NF + c) * HW + i0 + hi * 8;
#pragma unroll
    for (int r = 0; r < 8; ++r)
      out[base + r] = x[base + r] + al * (O[tt][r] * rv[r]);
  }
}

// ---------------------------------------------------------------------------
extern "C" void kernel_launch(void* const* d_in, const int* in_sizes, int n_in,
                              void* d_out, int out_size, void* d_ws, size_t ws_size,
                              hipStream_t stream) {
  const float* x     = (const float*)d_in[0];
  const float* bq    = (const float*)d_in[2];
  const float* Wq    = (const float*)d_in[1];
  const float* Wk    = (const float*)d_in[3];
  const float* bk    = (const float*)d_in[4];
  const float* Wv    = (const float*)d_in[5];
  const float* bv    = (const float*)d_in[6];
  const float* alpha = (const float*)d_in[7];
  float* out = (float*)d_out;

  // workspace layout (~20.2 MB)
  ushort_t* qt = (ushort_t*)d_ws;                 // [BS][HW][C8] bf16
  ushort_t* kt = qt + (size_t)BS * HW * C8;       // [BS][HW][C8] bf16
  ushort_t* vb = kt + (size_t)BS * HW * C8;       // [BS][NF][HW] bf16 (bias folded)
  float* mrow = (float*)(vb + (size_t)BS * NF * HW);  // [BS][HW]
  float* rinv = mrow + (size_t)BS * HW;               // [BS][HW]

  pam_proj_qk<<<dim3(HW / 256, 8, BS), 256, 0, stream>>>(x, Wq, bq, Wk, bk, qt, kt);
  pam_proj_v <<<dim3(HW / 256, NF / 16, BS), 256, 0, stream>>>(x, Wv, bv, vb);
  pam_stats  <<<dim3(HW, BS), 256, 0, stream>>>(qt, kt, mrow, rinv);
  pam_attn_out<<<dim3(HW / 16, BS), 256, 0, stream>>>(qt, kt, vb, mrow, rinv,
                                                      x, alpha, out);
}
